// GATLayer_9869834846959
// MI455X (gfx1250) — matrix operand order repaired and checked by reading, not compiled, not run on hardware
//
#include <hip/hip_runtime.h>
#include <hip/hip_bf16.h>
#include <stdint.h>

// ---------------------------------------------------------------------------
// GAT layer for MI455X (gfx1250, wave32).
//  h = x@W via v_wmma_f32_16x16x32_bf16 (f32 accumulate) with fragment-packed
//  LDS tiles (all ds traffic is b128), then edge-parallel segment softmax +
//  aggregation using L2-resident float atomics.
// ---------------------------------------------------------------------------

typedef __bf16          v16bf  __attribute__((ext_vector_type(16)));
typedef unsigned short  v8u16  __attribute__((ext_vector_type(8)));
typedef float           v8f    __attribute__((ext_vector_type(8)));

struct u16x16 { v8u16 lo, hi; };   // 32B, no padding -> bit_cast to v16bf

#define IN_F   256
#define HF     256   // HEADS * OUT_F
#define HEADS  4
#define OUT_F  64
#define NEG_SLOPE 0.2f

#define BM 128
#define BN 64
#define BK 32
#define ASTR 40   // LDS row stride (ushorts) for A tile  [BM][ASTR]
#define BSTR 40   // LDS col stride (ushorts) for B tile  [BN][BSTR] (frag-packed)

// ---- order-preserving float <-> uint key (for atomicMax-based segment max) --
__device__ __forceinline__ unsigned fkey(float f) {
    unsigned u = __float_as_uint(f);
    return (u & 0x80000000u) ? ~u : (u | 0x80000000u);
}
__device__ __forceinline__ float funkey(unsigned k) {
    unsigned u = (k & 0x80000000u) ? (k & 0x7FFFFFFFu) : ~k;
    return __uint_as_float(u);
}
__device__ __forceinline__ float lrelu(float x) { return x > 0.f ? x : NEG_SLOPE * x; }

__device__ __forceinline__ unsigned short f2bf(float f) {
    unsigned u = __float_as_uint(f);
    return (unsigned short)((u + 0x7FFFu + ((u >> 16) & 1u)) >> 16);
}

// ---------------------------------------------------------------------------
// fp32 -> bf16 bits (RNE)
// ---------------------------------------------------------------------------
__global__ __launch_bounds__(256) void gat_cvt_bf16(const float* __restrict__ src,
                                                    unsigned short* __restrict__ dst,
                                                    int n) {
    int i = blockIdx.x * blockDim.x + threadIdx.x;
    if (i >= n) return;
    dst[i] = f2bf(src[i]);
}

// W[k][n] (fp32) -> Wt[n][k] (bf16): puts K contiguous per output column so
// B fragments stage as straight b128 copies.
__global__ __launch_bounds__(256) void gat_cvt_wt(const float* __restrict__ W,
                                                  unsigned short* __restrict__ Wt) {
    int i = blockIdx.x * blockDim.x + threadIdx.x;
    if (i >= IN_F * HF) return;
    int k = i >> 8, n = i & 255;
    Wt[n * IN_F + k] = f2bf(W[i]);
}

// ---------------------------------------------------------------------------
// H[N,256] = X[N,256] @ W[256,256]  (bf16 in, f32 accumulate)
// block = 128 threads (4 waves); block tile 128x64; wave tile 32x64.
// ---------------------------------------------------------------------------
__global__ __launch_bounds__(128) void gat_gemm_wmma(const unsigned short* __restrict__ Xb,
                                                     const unsigned short* __restrict__ Wt,
                                                     float* __restrict__ H, int N) {
    __shared__ __align__(16) unsigned short As[BM * ASTR];  // row-major rows of X
    __shared__ __align__(16) unsigned short Bs[BN * BSTR];  // frag-packed: [col][k]

    const int tid  = threadIdx.x;
    const int lane = tid & 31;
    const int wave = tid >> 5;
    const int row0 = blockIdx.x * BM;
    const int col0 = blockIdx.y * BN;

    v8f acc[2][4] = {};

    // bf16 fragment k order used by this code (wave32, contiguous; A and B match):
    //  A 16x32: lane<16 holds K {0..15}, lane>=16 holds K {16..31}
    //  B 32x16: lane = N column; slot t <-> K = kb + t,  kb = (lane<16)?0:16
    const int mrow = lane & 15;
    const int kA8  = (lane < 16) ? 0 : 2;   // kbaseA/8  (A 16B-chunk select)
    const int kB8  = (lane < 16) ? 0 : 2;   // kbaseB/8  (B 16B-chunk select)
    const int ncol = lane & 15;

    const v8u16* Ap = (const v8u16*)As;     // ASTR/8 == 5 chunks per row
    const v8u16* Bp = (const v8u16*)Bs;     // BSTR/8 == 5 chunks per col

    for (int k0 = 0; k0 < IN_F; k0 += BK) {
        // ---- stage A tile: thread t copies row row0+t, 32 bf16 = 4 x b128 ----
        {
            uint4 v0 = {}, v1 = {}, v2 = {}, v3 = {};
            if (row0 + tid < N) {
                const uint4* src = (const uint4*)(Xb + (size_t)(row0 + tid) * IN_F + k0);
                v0 = src[0]; v1 = src[1]; v2 = src[2]; v3 = src[3];
            }
            uint4* dst = (uint4*)(As + tid * ASTR);
            dst[0] = v0; dst[1] = v1; dst[2] = v2; dst[3] = v3;
        }
        // ---- stage B tile (frag-packed): thread t copies half of Wt row ----
        {
            const int n    = tid >> 1;
            const int half = (tid & 1) << 4;   // 0 or 16 ushorts
            const uint4* src = (const uint4*)(Wt + (size_t)(col0 + n) * IN_F + k0 + half);
            uint4 v0 = src[0], v1 = src[1];
            uint4* dst = (uint4*)(Bs + n * BSTR + half);
            dst[0] = v0; dst[1] = v1;
        }
        // prefetch next K-tile into GL2 (global_prefetch_b8)
        if (k0 + BK < IN_F) {
            if (row0 + tid < N)
                __builtin_prefetch(Xb + (size_t)(row0 + tid) * IN_F + k0 + BK, 0, 1);
            __builtin_prefetch(Wt + (size_t)(col0 + (tid >> 1)) * IN_F + k0 + BK
                                   + ((tid & 1) << 4), 0, 1);
        }
        __syncthreads();

        // ---- B fragments: 2 x ds_load_b128 each, shared by both M sub-tiles ----
        v16bf bfrag[4];
        #pragma unroll
        for (int nf = 0; nf < 4; ++nf) {
            const int bc = nf * 16 + ncol;
            u16x16 t;
            t.lo = Bp[bc * (BSTR / 8) + kB8];
            t.hi = Bp[bc * (BSTR / 8) + kB8 + 1];
            bfrag[nf] = __builtin_bit_cast(v16bf, t);
        }
        // ---- 2 A fragments x 4 B fragments = 8 WMMA per wave per K-step ----
        #pragma unroll
        for (int m = 0; m < 2; ++m) {
            const int ar = wave * 32 + m * 16 + mrow;
            u16x16 t;
            t.lo = Ap[ar * (ASTR / 8) + kA8];
            t.hi = Ap[ar * (ASTR / 8) + kA8 + 1];
            v16bf afrag = __builtin_bit_cast(v16bf, t);
            #pragma unroll
            for (int nf = 0; nf < 4; ++nf)
                acc[m][nf] = __builtin_amdgcn_wmma_f32_16x16x32_bf16(
                    false, afrag, false, bfrag[nf], (short)0, acc[m][nf], false, false);
        }
        __syncthreads();
    }

    // ---- store: C 16x16 f32 layout: lanes 0-15 M=r, lanes 16-31 M=r+8, N=lane&15
    #pragma unroll
    for (int m = 0; m < 2; ++m) {
        const int rbase = row0 + wave * 32 + m * 16 + ((lane < 16) ? 0 : 8);
        #pragma unroll
        for (int nf = 0; nf < 4; ++nf) {
            const int gcol = col0 + nf * 16 + ncol;
            #pragma unroll
            for (int r = 0; r < 8; ++r)
                if (rbase + r < N) H[(size_t)(rbase + r) * HF + gcol] = acc[m][nf][r];
        }
    }
}

// ---------------------------------------------------------------------------
// wave per (node,head): a_src/a_dst dot products + seed e_max with self-loop
// ---------------------------------------------------------------------------
__global__ __launch_bounds__(256) void gat_node_att(const float* __restrict__ H,
                                                    const float* __restrict__ att_s,
                                                    const float* __restrict__ att_d,
                                                    float* __restrict__ a_src,
                                                    float* __restrict__ a_dst,
                                                    unsigned* __restrict__ emaxk, int N) {
    int gw = blockIdx.x * (blockDim.x >> 5) + (threadIdx.x >> 5);
    if (gw >= N * HEADS) return;
    int node = gw >> 2, head = gw & 3, lane = threadIdx.x & 31;
    const float* hp = H + (size_t)node * HF + head * OUT_F;
    float h0 = hp[lane], h1 = hp[lane + 32];
    float s = h0 * att_s[head * OUT_F + lane] + h1 * att_s[head * OUT_F + lane + 32];
    float d = h0 * att_d[head * OUT_F + lane] + h1 * att_d[head * OUT_F + lane + 32];
    #pragma unroll
    for (int off = 16; off > 0; off >>= 1) {
        s += __shfl_xor(s, off, 32);
        d += __shfl_xor(d, off, 32);
    }
    if (lane == 0) {
        a_src[gw] = s;
        a_dst[gw] = d;
        emaxk[gw] = fkey(lrelu(s + d));   // self-loop always exists -> finite max
    }
}

// ---------------------------------------------------------------------------
// thread per (edge,head): segment max via order-preserving uint atomicMax
// ---------------------------------------------------------------------------
__global__ __launch_bounds__(256) void gat_edge_max(const int* __restrict__ ei,
                                                    const float* __restrict__ a_src,
                                                    const float* __restrict__ a_dst,
                                                    unsigned* __restrict__ emaxk, int E) {
    int i = blockIdx.x * blockDim.x + threadIdx.x;
    if (i >= E * HEADS) return;
    int e = i >> 2, head = i & 3;
    int s = ei[e], d = ei[E + e];
    float v = lrelu(a_src[s * HEADS + head] + a_dst[d * HEADS + head]);
    atomicMax(&emaxk[d * HEADS + head], fkey(v));
}

// ---------------------------------------------------------------------------
// wave per (node,head): self-loop term initializes denom and out (plain stores,
// first writer by stream order — no memset needed)
// ---------------------------------------------------------------------------
__global__ __launch_bounds__(256) void gat_node_self(const float* __restrict__ H,
                                                     const float* __restrict__ a_src,
                                                     const float* __restrict__ a_dst,
                                                     const unsigned* __restrict__ emaxk,
                                                     float* __restrict__ denom,
                                                     float* __restrict__ out, int N) {
    int gw = blockIdx.x * (blockDim.x >> 5) + (threadIdx.x >> 5);
    if (gw >= N * HEADS) return;
    int node = gw >> 2, head = gw & 3, lane = threadIdx.x & 31;
    float e = lrelu(a_src[gw] + a_dst[gw]);
    float w = __expf(e - funkey(emaxk[gw]));
    if (lane == 0) denom[gw] = w;
    const float* hp = H + (size_t)node * HF + head * OUT_F;
    float* op = out + (size_t)node * HF + head * OUT_F;
    op[lane]      = w * hp[lane];
    op[lane + 32] = w * hp[lane + 32];
}

// ---------------------------------------------------------------------------
// wave per edge: unnormalized weighted aggregation (exp + fused gather/scatter)
// out[dst] += w * h[src]; denom[dst] += w.  Divide in finalize.
// ---------------------------------------------------------------------------
__global__ __launch_bounds__(256) void gat_edge_accum(const int* __restrict__ ei,
                                                      const float* __restrict__ a_src,
                                                      const float* __restrict__ a_dst,
                                                      const unsigned* __restrict__ emaxk,
                                                      const float* __restrict__ H,
                                                      float* __restrict__ denom,
                                                      float* __restrict__ out, int E) {
    int gw = blockIdx.x * (blockDim.x >> 5) + (threadIdx.x >> 5);
    if (gw >= E) return;
    int lane = threadIdx.x & 31;
    int s = ei[gw], d = ei[E + gw];
    const float* hs = H + (size_t)s * HF;
    float* od = out + (size_t)d * HF;
    #pragma unroll
    for (int head = 0; head < HEADS; ++head) {
        float e = lrelu(a_src[s * HEADS + head] + a_dst[d * HEADS + head]);
        float w = __expf(e - funkey(emaxk[d * HEADS + head]));
        if (lane == 0) atomicAdd(&denom[d * HEADS + head], w);
        atomicAdd(&od[head * OUT_F + lane],      w * hs[head * OUT_F + lane]);
        atomicAdd(&od[head * OUT_F + lane + 32], w * hs[head * OUT_F + lane + 32]);
    }
}

// ---------------------------------------------------------------------------
// out = out / denom + bias
// ---------------------------------------------------------------------------
__global__ __launch_bounds__(256) void gat_finalize(float* __restrict__ out,
                                                    const float* __restrict__ denom,
                                                    const float* __restrict__ bias, int N) {
    int i = blockIdx.x * blockDim.x + threadIdx.x;
    if (i >= N * HF) return;
    int n = i >> 8, c = i & 255, head = c >> 6;
    out[i] = out[i] / denom[n * HEADS + head] + bias[c];
}

// ---------------------------------------------------------------------------
extern "C" void kernel_launch(void* const* d_in, const int* in_sizes, int n_in,
                              void* d_out, int out_size, void* d_ws, size_t ws_size,
                              hipStream_t stream) {
    const float* x     = (const float*)d_in[0];
    const int*   ei    = (const int*)d_in[1];
    const float* W     = (const float*)d_in[2];
    const float* att_s = (const float*)d_in[3];
    const float* att_d = (const float*)d_in[4];
    const float* bias  = (const float*)d_in[5];
    float*       out   = (float*)d_out;

    const int N = in_sizes[0] / IN_F;
    const int E = in_sizes[1] / 2;

    // workspace carve-up (~80 MB total)
    char* ws = (char*)d_ws;
    unsigned short* xb = (unsigned short*)ws;                       // N*256 bf16
    unsigned short* wt = xb + (size_t)N * IN_F;                     // 256*256 bf16 (transposed)
    float* H     = (float*)(wt + (size_t)IN_F * HF);                // N*256 f32
    float* a_src = H + (size_t)N * HF;                              // N*4
    float* a_dst = a_src + (size_t)N * HEADS;                       // N*4
    unsigned* emaxk = (unsigned*)(a_dst + (size_t)N * HEADS);       // N*4
    float* denom = (float*)(emaxk + (size_t)N * HEADS);             // N*4

    gat_cvt_bf16<<<(N * IN_F + 255) / 256, 256, 0, stream>>>(x, xb, N * IN_F);
    gat_cvt_wt<<<(IN_F * HF + 255) / 256, 256, 0, stream>>>(W, wt);

    dim3 ggrid((N + BM - 1) / BM, HF / BN);
    gat_gemm_wmma<<<ggrid, 128, 0, stream>>>(xb, wt, H, N);

    gat_node_att<<<(N * HEADS + 7) / 8, 256, 0, stream>>>(H, att_s, att_d,
                                                          a_src, a_dst, emaxk, N);
    gat_edge_max<<<(E * HEADS + 255) / 256, 256, 0, stream>>>(ei, a_src, a_dst, emaxk, E);
    gat_node_self<<<(N * HEADS + 7) / 8, 256, 0, stream>>>(H, a_src, a_dst, emaxk,
                                                           denom, out, N);
    gat_edge_accum<<<(E + 7) / 8, 256, 0, stream>>>(ei, a_src, a_dst, emaxk, H,
                                                    denom, out, E);
    gat_finalize<<<(N * HF + 255) / 256, 256, 0, stream>>>(out, denom, bias, N);
}
